// GATSBMs_15479062135606
// MI455X (gfx1250) — matrix operand order repaired and checked by reading, not compiled, not run on hardware
//
#include <hip/hip_runtime.h>
#include <hip/hip_bf16.h>

// ---------------------------------------------------------------------------
// GAT (3 layers, H=4, D=32, HD=128) + BN + 2-layer classifier for gfx1250.
// Dense projections use V_WMMA_F32_16X16X32_BF16. Operands are pre-swizzled
// into fragment-ordered bf16 buffers so each lane's 16-element fragment is a
// single contiguous 32B load. One wave computes a 16x128 strip (32 WMMAs).
// Edge softmax + aggregation use L2-resident gathers and float atomics.
// ---------------------------------------------------------------------------

#define HD 128
#define NH 4
#define DH 32
#define NEG_SLOPE 0.2f
#define BN_EPS 1e-5f

typedef __attribute__((ext_vector_type(16))) __bf16 v16bf;
typedef __attribute__((ext_vector_type(8)))  float  v8f;

static __device__ __forceinline__ unsigned short f2bf_u16(float f) {
    unsigned u = __builtin_bit_cast(unsigned, f);
    unsigned r = u + 0x7FFFu + ((u >> 16) & 1u);   // round-to-nearest-even
    return (unsigned short)(r >> 16);
}

static __device__ __forceinline__ unsigned enc_ord(float f) {
    unsigned u = __builtin_bit_cast(unsigned, f);
    return (u & 0x80000000u) ? ~u : (u | 0x80000000u);   // monotonic float->uint
}
static __device__ __forceinline__ float dec_ord(unsigned u) {
    unsigned b = (u & 0x80000000u) ? (u ^ 0x80000000u) : ~u;
    return __builtin_bit_cast(float, b);
}
static __device__ __forceinline__ float elu1(float v) {
    return v > 0.0f ? v : (expf(v) - 1.0f);
}

// ---------------------------------------------------------------------------
__global__ void k_fill_u32(unsigned* __restrict__ p, unsigned v, int n) {
    int i = blockIdx.x * blockDim.x + threadIdx.x;
    if (i < n) p[i] = v;
}

// embedding lookup: h[n][c] = embed[x[n]][c]
__global__ void k_embed(const int* __restrict__ x, const float* __restrict__ embed,
                        float* __restrict__ h, int n_nodes) {
    int i = blockIdx.x * blockDim.x + threadIdx.x;
    if (i >= n_nodes * HD) return;
    int n = i >> 7, c = i & 127;
    h[i] = embed[x[n] * HD + c];
}

// ---------------------------------------------------------------------------
// Swizzle h (f32, row-major [N][128]) into fragment-ordered bf16:
//   Ab[row*128 + ks*32 + grp*16 + q]  with K = ks*32 + grp*8 + (q<8 ? q : q+8)
// so a lane's full 16x32 A-fragment slice is 16 contiguous bf16 (32 bytes).
__global__ void k_h2bf_swz(const float* __restrict__ h, unsigned short* __restrict__ Ab,
                           int n_nodes) {
    int i = blockIdx.x * blockDim.x + threadIdx.x;
    if (i >= n_nodes * HD) return;
    int row = i >> 7, j = i & 127;
    int ks = j >> 5, r = j & 31, grp = (r >> 4) & 1, q = r & 15;
    int K = ks * 32 + grp * 8 + (q < 8 ? q : q + 8);
    Ab[i] = f2bf_u16(h[row * HD + K]);
}

// Swizzle W (f32, [128 x ncols]) into fragment-ordered bf16:
//   Wb[((nt*4 + ks)*32 + lane)*16 + i] = W[(ks*32 + (lane>>4)*8 + (i<8 ? i : i+8))*ncols + nt*16 + lane&15]
__global__ void k_w2bf_swz(const float* __restrict__ W, unsigned short* __restrict__ Wb,
                           int ncols) {
    int t = blockIdx.x * blockDim.x + threadIdx.x;
    if (t >= HD * ncols) return;
    int i = t & 15, lane = (t >> 4) & 31, ks = (t >> 9) & 3, nt = t >> 11;
    int grp = lane >> 4, lm = lane & 15;
    int K = ks * 32 + grp * 8 + (i < 8 ? i : i + 8);
    Wb[t] = f2bf_u16(W[K * ncols + nt * 16 + lm]);
}

// ---------------------------------------------------------------------------
// C[M x 128] = A[M x 128] @ B[128 x 128] from pre-swizzled bf16 operands.
// One wave -> one 16x128 strip: A fragment loaded once per k-step and reused
// across 8 n-tiles; 32 WMMAs per wave. Block (32,4) -> 4 strips per block.
__global__ void k_gemm_wmma(const unsigned short* __restrict__ Ab,
                            const unsigned short* __restrict__ Wb,
                            float* __restrict__ C, int n_rows) {
    const int lane = threadIdx.x;          // 0..31
    const int grp  = lane >> 4;
    const int lm   = lane & 15;
    const int m0   = (blockIdx.x * 4 + threadIdx.y) * 16;
    int row = m0 + lm;
    if (row >= n_rows) row = n_rows - 1;   // clamp (keeps EXEC all-ones)

    v8f acc[8] = {};
#pragma unroll
    for (int ks = 0; ks < 4; ++ks) {
        const v16bf a = *(const v16bf*)&Ab[row * HD + ks * 32 + grp * 16];
#pragma unroll
        for (int nt = 0; nt < 8; ++nt) {
            const v16bf b = *(const v16bf*)&Wb[((nt * 4 + ks) * 32 + lane) * 16];
            acc[nt] = __builtin_amdgcn_wmma_f32_16x16x32_bf16(
                false, a, false, b, (short)0, acc[nt], false, false);
        }
    }
#pragma unroll
    for (int nt = 0; nt < 8; ++nt) {
#pragma unroll
        for (int r = 0; r < 8; ++r) {
            const int m = m0 + grp * 8 + r;
            if (m < n_rows) C[m * HD + nt * 16 + lm] = acc[nt][r];
        }
    }
}

// t1[M x 64] = relu(A[M x 128] @ W1[128 x 64] + b1), WMMA version (4 n-tiles)
__global__ void k_cls1_wmma(const unsigned short* __restrict__ Ab,
                            const unsigned short* __restrict__ Wb,
                            const float* __restrict__ bias,
                            float* __restrict__ t1, int n_rows) {
    const int lane = threadIdx.x;
    const int grp  = lane >> 4;
    const int lm   = lane & 15;
    const int m0   = (blockIdx.x * 4 + threadIdx.y) * 16;
    int row = m0 + lm;
    if (row >= n_rows) row = n_rows - 1;

    v8f acc[4] = {};
#pragma unroll
    for (int ks = 0; ks < 4; ++ks) {
        const v16bf a = *(const v16bf*)&Ab[row * HD + ks * 32 + grp * 16];
#pragma unroll
        for (int nt = 0; nt < 4; ++nt) {
            const v16bf b = *(const v16bf*)&Wb[((nt * 4 + ks) * 32 + lane) * 16];
            acc[nt] = __builtin_amdgcn_wmma_f32_16x16x32_bf16(
                false, a, false, b, (short)0, acc[nt], false, false);
        }
    }
#pragma unroll
    for (int nt = 0; nt < 4; ++nt) {
        const int col = nt * 16 + lm;
        const float bc = bias[col];
#pragma unroll
        for (int r = 0; r < 8; ++r) {
            const int m = m0 + grp * 8 + r;
            if (m < n_rows) {
                float v = acc[nt][r] + bc;
                t1[m * 64 + col] = v > 0.f ? v : 0.f;
            }
        }
    }
}

// ---------------------------------------------------------------------------
// el[n][h] = sum_d feat[n][h][d]*al[h][d] ; er likewise
__global__ void k_attn(const float* __restrict__ feat, const float* __restrict__ al,
                       const float* __restrict__ ar, float* __restrict__ el,
                       float* __restrict__ er, int n_nodes) {
    int t = blockIdx.x * blockDim.x + threadIdx.x;
    if (t >= n_nodes * NH) return;
    int n = t >> 2, h = t & 3;
    float sl = 0.f, sr = 0.f;
#pragma unroll
    for (int d = 0; d < DH; ++d) {
        float f = feat[n * HD + h * DH + d];
        sl += f * al[h * DH + d];
        sr += f * ar[h * DH + d];
    }
    el[t] = sl;
    er[t] = sr;
}

// e = leaky_relu(el[src]+er[dst]); segment max over dst via ordered-uint atomicMax
__global__ void k_edge_max(const int* __restrict__ src, const int* __restrict__ dst,
                           const float* __restrict__ el, const float* __restrict__ er,
                           float* __restrict__ escore, unsigned* __restrict__ menc,
                           int n_edges) {
    int e = blockIdx.x * blockDim.x + threadIdx.x;
    if (e >= n_edges) return;
    int s = src[e], d = dst[e];
#pragma unroll
    for (int h = 0; h < NH; ++h) {
        float v = el[s * NH + h] + er[d * NH + h];
        v = v > 0.0f ? v : NEG_SLOPE * v;
        escore[e * NH + h] = v;
        atomicMax(&menc[d * NH + h], enc_ord(v));
    }
}

// ee = exp(e - m[dst]); s[dst] += ee
__global__ void k_edge_expsum(const int* __restrict__ dst, float* __restrict__ escore,
                              const unsigned* __restrict__ menc, float* __restrict__ ssum,
                              int n_edges) {
    int e = blockIdx.x * blockDim.x + threadIdx.x;
    if (e >= n_edges) return;
    int d = dst[e];
#pragma unroll
    for (int h = 0; h < NH; ++h) {
        float m  = dec_ord(menc[d * NH + h]);
        float ee = expf(escore[e * NH + h] - m);
        escore[e * NH + h] = ee;
        atomicAdd(&ssum[d * NH + h], ee);
    }
}

// rst[dst] += feat[src] * (ee / s[dst]) ; one thread per (edge, float4 chunk)
__global__ void k_aggregate(const int* __restrict__ src, const int* __restrict__ dst,
                            const float* __restrict__ feat, const float* __restrict__ escore,
                            const float* __restrict__ ssum, float* __restrict__ rst,
                            int n_edges) {
    long t = (long)blockIdx.x * blockDim.x + threadIdx.x;
    if (t >= (long)n_edges * 32) return;
    int e = (int)(t >> 5);
    int q = (int)(t & 31);            // 4-float chunk index within the 128 features
    int s = src[e], d = dst[e];
    int head = q >> 3;                // 32 floats per head -> 8 chunks per head
    float a = escore[e * NH + head] / ssum[d * NH + head];
    const float4 f = *(const float4*)&feat[s * HD + q * 4];
    float* o = &rst[d * HD + q * 4];
    atomicAdd(o + 0, f.x * a);
    atomicAdd(o + 1, f.y * a);
    atomicAdd(o + 2, f.z * a);
    atomicAdd(o + 3, f.w * a);
}

// rst = (rst [+ h_in]) [elu] * snorm_n ; accumulate per-column sum / sumsq
// block = 128 threads (one per column), each block covers 64 rows
__global__ void k_post(float* __restrict__ rst, const float* __restrict__ h_in,
                       const float* __restrict__ snorm_n, float* __restrict__ colsum,
                       float* __restrict__ colsum2, int residual, int act, int n_nodes) {
    const int c  = threadIdx.x;
    const int r0 = blockIdx.x * 64;
    float s1 = 0.f, s2 = 0.f;
    for (int i = 0; i < 64; ++i) {
        int n = r0 + i;
        if (n >= n_nodes) break;
        float v = rst[n * HD + c];
        if (residual) v += h_in[n * HD + c];
        if (act) v = elu1(v);
        v *= snorm_n[n];
        rst[n * HD + c] = v;
        s1 += v;
        s2 += v * v;
    }
    atomicAdd(&colsum[c], s1);
    atomicAdd(&colsum2[c], s2);
}

__global__ void k_bn_finalize(const float* __restrict__ colsum, const float* __restrict__ colsum2,
                              float* __restrict__ mu, float* __restrict__ istd, int n_nodes) {
    int c = threadIdx.x;
    float m = colsum[c] / (float)n_nodes;
    float v = colsum2[c] / (float)n_nodes - m * m;
    mu[c] = m;
    istd[c] = rsqrtf(v + BN_EPS);
}

// h_out = elu(gamma*(h_mid - mu)*istd + beta)
__global__ void k_bn_elu(const float* __restrict__ h_mid, const float* __restrict__ gamma,
                         const float* __restrict__ beta, const float* __restrict__ mu,
                         const float* __restrict__ istd, float* __restrict__ h_out,
                         int n_nodes) {
    int i = blockIdx.x * blockDim.x + threadIdx.x;
    if (i >= n_nodes * HD) return;
    int c = i & 127;
    float v = gamma[c] * (h_mid[i] - mu[c]) * istd[c] + beta[c];
    h_out[i] = elu1(v);
}

// classifier stage 2: out[n][0..1] = t1 @ W2 + b2, W2: [64 x 2]
__global__ void k_cls2(const float* __restrict__ t1, const float* __restrict__ w,
                       const float* __restrict__ b, float* __restrict__ out, int n_nodes) {
    int n = blockIdx.x * blockDim.x + threadIdx.x;
    if (n >= n_nodes) return;
    float s0 = b[0], s1 = b[1];
#pragma unroll 8
    for (int k = 0; k < 64; ++k) {
        float v = t1[n * 64 + k];
        s0 += v * w[k * 2 + 0];
        s1 += v * w[k * 2 + 1];
    }
    out[n * 2 + 0] = s0;
    out[n * 2 + 1] = s1;
}

// ---------------------------------------------------------------------------
static inline int cdiv(long a, long b) { return (int)((a + b - 1) / b); }

extern "C" void kernel_launch(void* const* d_in, const int* in_sizes, int n_in,
                              void* d_out, int out_size, void* d_ws, size_t ws_size,
                              hipStream_t stream) {
    const int n_nodes = in_sizes[0];
    const int n_edges = in_sizes[1];

    const int*   x       = (const int*)d_in[0];
    const int*   src     = (const int*)d_in[1];
    const int*   dst     = (const int*)d_in[2];
    const float* snorm_n = (const float*)d_in[3];
    // d_in[4] = snorm_e (unused by the reference)
    const float* embed   = (const float*)d_in[5];
    const float* Ws[3]     = { (const float*)d_in[6],  (const float*)d_in[11], (const float*)d_in[16] };
    const float* als[3]    = { (const float*)d_in[7],  (const float*)d_in[12], (const float*)d_in[17] };
    const float* ars[3]    = { (const float*)d_in[8],  (const float*)d_in[13], (const float*)d_in[18] };
    const float* gammas[3] = { (const float*)d_in[9],  (const float*)d_in[14], (const float*)d_in[19] };
    const float* betas[3]  = { (const float*)d_in[10], (const float*)d_in[15], (const float*)d_in[20] };
    const float* cls1_w = (const float*)d_in[21];
    const float* cls1_b = (const float*)d_in[22];
    const float* cls2_w = (const float*)d_in[23];
    const float* cls2_b = (const float*)d_in[24];
    float* out = (float*)d_out;

    // ------- carve workspace -------
    char* ws = (char*)d_ws;
    size_t off = 0;
    auto carve = [&](size_t bytes) -> void* {
        off = (off + 255) & ~(size_t)255;
        void* p = ws + off;
        off += bytes;
        return p;
    };
    float*          hA      = (float*)carve((size_t)n_nodes * HD * 4);
    float*          hB      = (float*)carve((size_t)n_nodes * HD * 4);   // rst / h_mid
    float*          feat    = (float*)carve((size_t)n_nodes * HD * 4);
    unsigned short* Ab      = (unsigned short*)carve((size_t)n_nodes * HD * 2);
    unsigned short* Wb      = (unsigned short*)carve((size_t)HD * HD * 2);
    unsigned short* Wb1     = (unsigned short*)carve((size_t)HD * 64 * 2);
    float*          el      = (float*)carve((size_t)n_nodes * NH * 4);
    float*          er      = (float*)carve((size_t)n_nodes * NH * 4);
    float*          escore  = (float*)carve((size_t)n_edges * NH * 4);
    unsigned*       menc    = (unsigned*)carve((size_t)n_nodes * NH * 4);
    float*          ssum    = (float*)carve((size_t)n_nodes * NH * 4);
    float*          colsum  = (float*)carve(HD * 4);
    float*          colsum2 = (float*)carve(HD * 4);
    float*          mu      = (float*)carve(HD * 4);
    float*          istd    = (float*)carve(HD * 4);
    float*          t1      = (float*)carve((size_t)n_nodes * 64 * 4);
    (void)ws_size; (void)n_in; (void)out_size;

    const int B = 256;
    const dim3 gemm_grid(cdiv(n_nodes, 64));
    const dim3 gemm_block(32, 4);

    // h0 = embed[x]
    k_embed<<<cdiv((long)n_nodes * HD, B), B, 0, stream>>>(x, embed, hA, n_nodes);

    for (int layer = 0; layer < 3; ++layer) {
        const int residual = (layer > 0) ? 1 : 0;
        const int act      = (layer > 0) ? 1 : 0;

        // swizzle operands to bf16 fragment order, then feat = hA @ W via WMMA
        k_h2bf_swz<<<cdiv((long)n_nodes * HD, B), B, 0, stream>>>(hA, Ab, n_nodes);
        k_w2bf_swz<<<cdiv(HD * HD, B), B, 0, stream>>>(Ws[layer], Wb, HD);
        k_gemm_wmma<<<gemm_grid, gemm_block, 0, stream>>>(Ab, Wb, feat, n_nodes);

        // attention logits
        k_attn<<<cdiv((long)n_nodes * NH, B), B, 0, stream>>>(feat, als[layer], ars[layer],
                                                              el, er, n_nodes);

        // clear per-layer accumulators
        k_fill_u32<<<cdiv((long)n_nodes * NH, B), B, 0, stream>>>(menc, 0u, n_nodes * NH);
        k_fill_u32<<<cdiv((long)n_nodes * NH, B), B, 0, stream>>>((unsigned*)ssum, 0u, n_nodes * NH);
        k_fill_u32<<<cdiv((long)n_nodes * HD, B), B, 0, stream>>>((unsigned*)hB, 0u, n_nodes * HD);
        k_fill_u32<<<1, 128, 0, stream>>>((unsigned*)colsum, 0u, HD);
        k_fill_u32<<<1, 128, 0, stream>>>((unsigned*)colsum2, 0u, HD);

        // edge softmax
        k_edge_max<<<cdiv(n_edges, B), B, 0, stream>>>(src, dst, el, er, escore, menc, n_edges);
        k_edge_expsum<<<cdiv(n_edges, B), B, 0, stream>>>(dst, escore, menc, ssum, n_edges);

        // message aggregation
        k_aggregate<<<cdiv((long)n_edges * 32, B), B, 0, stream>>>(src, dst, feat, escore,
                                                                   ssum, hB, n_edges);

        // residual + (elu) + snorm + BN stats, then BN + elu into hA
        k_post<<<cdiv(n_nodes, 64), 128, 0, stream>>>(hB, hA, snorm_n, colsum, colsum2,
                                                      residual, act, n_nodes);
        k_bn_finalize<<<1, 128, 0, stream>>>(colsum, colsum2, mu, istd, n_nodes);
        k_bn_elu<<<cdiv((long)n_nodes * HD, B), B, 0, stream>>>(hB, gammas[layer], betas[layer],
                                                                mu, istd, hA, n_nodes);
    }

    // classifier: t1 = relu(hA @ cls1_w + b1) via WMMA, then 64->2 scalar
    k_h2bf_swz<<<cdiv((long)n_nodes * HD, B), B, 0, stream>>>(hA, Ab, n_nodes);
    k_w2bf_swz<<<cdiv(HD * 64, B), B, 0, stream>>>(cls1_w, Wb1, 64);
    k_cls1_wmma<<<gemm_grid, gemm_block, 0, stream>>>(Ab, Wb1, cls1_b, t1, n_nodes);
    k_cls2<<<cdiv(n_nodes, B), B, 0, stream>>>(t1, cls2_w, cls2_b, out, n_nodes);
}
